// CausalSelfAttention_31765578121466
// MI455X (gfx1250) — matrix operand order repaired and checked by reading, not compiled, not run on hardware
//
#include <hip/hip_runtime.h>

typedef __bf16 bf16;
typedef __attribute__((ext_vector_type(16))) __bf16 v16bf;
typedef __attribute__((ext_vector_type(8)))  float  v8f;

#define NSEQ 2048
#define DMODEL 1024
#define NHEAD 16
#define HD 64
#define BATCH 2
#define BH (BATCH * NHEAD)

// ---------------------------------------------------------------------------
// WMMA fragment helpers (layouts per CDNA5 ISA 7.12.2, wave32)
// ---------------------------------------------------------------------------

// A-matrix 16x32 bf16 fragment from a row-major LDS tile [row][k], stride ld.
// lanes 0-15: M=lane, VGPR j -> K pairs {0,2,4,6,16,18,20,22}[j]
// lanes 16-31: M=lane-16, same + 8
__device__ inline v16bf load_a_frag(const bf16* lds, int row0, int k0, int ld, int lane) {
    union { v16bf v; unsigned u[8]; } f;
    const int m  = lane & 15;
    const int kg = (lane >> 4) << 3;  // 0 or 8
    const bf16* base = lds + (row0 + m) * ld + k0 + kg;
#pragma unroll
    for (int j = 0; j < 8; ++j) {
        const int k = (j < 4) ? (2 * j) : (16 + 2 * (j - 4));
        f.u[j] = *(const unsigned*)(base + k);
    }
    return f.v;
}

// B-matrix 32x16 bf16 fragment from a row-major LDS tile [k][n], stride ld.
// lane holds column N=lane&15; element s -> K = 8*(lane>>4) + {0..7, 16..23}[s].
__device__ inline v16bf load_b_frag(const bf16* lds, int k0, int n0, int ld, int lane) {
    union { v16bf v; bf16 e[16]; } f;
    const int kg = (lane >> 4) << 3;
    const bf16* base = lds + k0 * ld + n0 + (lane & 15);
#pragma unroll
    for (int s = 0; s < 16; ++s) {
        const int k = (s < 8) ? (kg + s) : (16 + kg + (s - 8));
        f.e[s] = base[k * ld];
    }
    return f.v;
}

__device__ inline v8f wmma_bf16(v16bf a, v16bf b, v8f c) {
    return __builtin_amdgcn_wmma_f32_16x16x32_bf16(false, a, false, b, (short)0, c,
                                                   false, false);
}

// CDNA5 async global->LDS copy: 16 bytes per lane, tracked by ASYNCcnt.
// (Inline asm: portable across ROCm-7.2 / amdgpu-toolchain, whose builtin
// arities differ.)
__device__ inline void async_copy_b128(unsigned lds_byte_addr, const void* gaddr) {
    asm volatile("global_load_async_to_lds_b128 %0, %1, off"
                 :: "v"(lds_byte_addr), "v"(gaddr) : "memory");
}
__device__ inline void wait_asynccnt0() {
    asm volatile("s_wait_asynccnt 0" ::: "memory");
}
__device__ inline unsigned lds_addr32(const void* p) {
    // low 32 bits of an LDS-aperture flat address are the LDS byte address
    return (unsigned)(unsigned long long)p;
}

// ---------------------------------------------------------------------------
// Kernel 1: per-head fused QKV projection (bias folded, Q pre-scaled 1/8)
//   x[b,n,d] (f32) @ Wqkv[h,d,0:192] -> K|Q|V bf16 tiles [bh][n][64]
// ---------------------------------------------------------------------------
__global__ __launch_bounds__(256)
void qkv_kernel(const float* __restrict__ x, const float* __restrict__ Wqkv,
                const float* __restrict__ bqkv,
                bf16* __restrict__ Kb, bf16* __restrict__ Qb, bf16* __restrict__ Vb) {
    const int bh = blockIdx.x;            // 0..31
    const int b = bh >> 4, h = bh & 15;
    const int m0 = blockIdx.y * 64;       // row tile
    const int tid = threadIdx.x;
    const int lane = tid & 31, wave = tid >> 5;
    const int strip = wave & 3;           // 16-row strip within the 64-row tile
    const int half  = wave >> 2;          // column half (0: cols 0-95, 1: 96-191)

    __shared__ bf16 Xs[64 * 32];
    __shared__ bf16 Ws[32 * 192];

    v8f acc[6];
#pragma unroll
    for (int t = 0; t < 6; ++t) acc[t] = (v8f){0, 0, 0, 0, 0, 0, 0, 0};

    const float* xb = x + (size_t)b * NSEQ * DMODEL;
    const float* Wh = Wqkv + (size_t)h * DMODEL * 192;

    for (int kk = 0; kk < DMODEL; kk += 32) {
        __builtin_prefetch(Wh + (size_t)(kk + 32) * 192, 0, 0);
        for (int idx = tid; idx < 64 * 32; idx += 256) {
            const int r = idx >> 5, c = idx & 31;
            Xs[idx] = (bf16)xb[(size_t)(m0 + r) * DMODEL + kk + c];
        }
        for (int idx = tid; idx < 32 * 192; idx += 256) {
            const int r = idx / 192, c = idx % 192;
            Ws[idx] = (bf16)Wh[(size_t)(kk + r) * 192 + c];
        }
        __syncthreads();
        const v16bf a = load_a_frag(Xs, strip * 16, 0, 32, lane);
#pragma unroll
        for (int t = 0; t < 6; ++t) {
            const v16bf bf = load_b_frag(Ws, 0, half * 96 + t * 16, 192, lane);
            acc[t] = wmma_bf16(a, bf, acc[t]);
        }
        __syncthreads();
    }

    // epilogue: C-layout scatter (VGPR r -> row r / r+8; lane&15 -> col)
    const int rowhi = (lane >> 4) << 3;
#pragma unroll
    for (int t = 0; t < 6; ++t) {
        const int col  = half * 96 + t * 16 + (lane & 15);
        const float bias = bqkv[h * 192 + col];
#pragma unroll
        for (int r = 0; r < 8; ++r) {
            const int row = m0 + strip * 16 + rowhi + r;
            const float v = acc[t][r] + bias;
            const size_t o = ((size_t)bh * NSEQ + row) * HD;
            if (col < 64)        Kb[o + col]         = (bf16)v;
            else if (col < 128)  Qb[o + (col - 64)]  = (bf16)(v * 0.125f);  // 1/sqrt(64)
            else                 Vb[o + (col - 128)] = (bf16)v;
        }
    }
}

// ---------------------------------------------------------------------------
// Kernel 2: causal flash attention, one (b,h) x 64-query tile per block
// 4 waves; each wave owns 16 query rows. Online softmax, bf16 WMMA.
// V tile staged with GLOBAL_LOAD_ASYNC_TO_LDS_B128 (ASYNCcnt); K tile staged
// with b128 global loads + transposed ds stores (so K^T B-frags are contiguous).
// ---------------------------------------------------------------------------
__global__ __launch_bounds__(128)
void flash_kernel(const bf16* __restrict__ Qb, const bf16* __restrict__ Kb,
                  const bf16* __restrict__ Vb, bf16* __restrict__ SA) {
    const int bh = blockIdx.x;
    const int b = bh >> 4, h = bh & 15;
    const int qt = blockIdx.y;
    const int q0 = qt * 64;
    const int tid = threadIdx.x, lane = tid & 31, wave = tid >> 5;

    __shared__ bf16 Qs[64 * 64];      // [row][hd]
    __shared__ bf16 Kt[64 * 64];      // transposed: [hd][key]
    __shared__ bf16 Vs[64 * 64];      // [key][hd]
    __shared__ bf16 Ps[4][16 * 72];   // per-wave P tile, padded stride 72

    const bf16* Qg = Qb + ((size_t)bh * NSEQ + q0) * HD;
    const bf16* Kg = Kb + (size_t)bh * NSEQ * HD;
    const bf16* Vg = Vb + (size_t)bh * NSEQ * HD;

    // Q tile: 16B-vector copy (8 KB / block -> 4 uint4 per thread)
    {
        const uint4* src = (const uint4*)Qg;
        uint4* dst = (uint4*)Qs;
        for (int i = tid; i < 512; i += 128) dst[i] = src[i];
    }
    __syncthreads();

    v16bf qa[2];
    qa[0] = load_a_frag(Qs, wave * 16, 0, 64, lane);
    qa[1] = load_a_frag(Qs, wave * 16, 32, 64, lane);

    float m[8], lsum[8];
    v8f o[4];
#pragma unroll
    for (int r = 0; r < 8; ++r) { m[r] = -3.0e30f; lsum[r] = 0.0f; }
#pragma unroll
    for (int d = 0; d < 4; ++d) o[d] = (v8f){0, 0, 0, 0, 0, 0, 0, 0};

    const int rowhi = (lane >> 4) << 3;
    const int myrow = q0 + wave * 16 + rowhi;  // +r per accumulator slot
    const unsigned vs_base = lds_addr32(Vs);

    for (int kt = 0; kt <= qt; ++kt) {
        const int kbase = kt * 64;
        __syncthreads();  // protect K/V LDS from readers of previous tile

        // V tile: async global->LDS, natural [key][hd] layout
        {
            const char* gv = (const char*)(Vg + (size_t)kbase * HD);
            for (int i = tid; i < 512; i += 128)
                async_copy_b128(vs_base + (unsigned)i * 16u, gv + (size_t)i * 16);
        }
        // K tile: b128 loads, transpose into Kt[hd][key]
        {
            const uint4* gk = (const uint4*)(Kg + (size_t)kbase * HD);
            for (int i = tid; i < 512; i += 128) {
                union { uint4 u; bf16 hh[8]; } t;
                t.u = gk[i];
                const int key = i >> 3, d0 = (i & 7) * 8;
#pragma unroll
                for (int j = 0; j < 8; ++j) Kt[(d0 + j) * 64 + key] = t.hh[j];
            }
        }
        __builtin_prefetch(Kg + (size_t)(kbase + 64) * HD, 0, 0);
        wait_asynccnt0();   // this thread's async copies have landed in LDS
        __syncthreads();    // ... and everyone else's

        // S = Q @ K^T  (contraction over hd=64 -> 2 WMMAs per 16x16 tile)
        v8f s[4];
#pragma unroll
        for (int nt = 0; nt < 4; ++nt) {
            v8f a = (v8f){0, 0, 0, 0, 0, 0, 0, 0};
            a = wmma_bf16(qa[0], load_b_frag(Kt, 0, nt * 16, 64, lane), a);
            a = wmma_bf16(qa[1], load_b_frag(Kt, 32, nt * 16, 64, lane), a);
            s[nt] = a;
        }

        if (kt == qt) {  // causal mask on the diagonal tile
#pragma unroll
            for (int nt = 0; nt < 4; ++nt) {
                const int colk = kbase + nt * 16 + (lane & 15);
#pragma unroll
                for (int r = 0; r < 8; ++r)
                    if (colk > myrow + r) s[nt][r] = -3.0e30f;
            }
        }

        // online softmax: rows live across 16-lane halves in the C layout
        float nm[8];
#pragma unroll
        for (int r = 0; r < 8; ++r) {
            float mx = m[r];
#pragma unroll
            for (int nt = 0; nt < 4; ++nt) mx = fmaxf(mx, s[nt][r]);
#pragma unroll
            for (int d = 1; d < 16; d <<= 1) mx = fmaxf(mx, __shfl_xor(mx, d, 32));
            nm[r] = mx;
        }
        float psum[8];
#pragma unroll
        for (int r = 0; r < 8; ++r) {
            const float alpha = __expf(m[r] - nm[r]);
            m[r] = nm[r];
            lsum[r] *= alpha;
#pragma unroll
            for (int d = 0; d < 4; ++d) o[d][r] *= alpha;
            psum[r] = 0.0f;
        }
        bf16* Pw = &Ps[wave][0];
#pragma unroll
        for (int nt = 0; nt < 4; ++nt) {
#pragma unroll
            for (int r = 0; r < 8; ++r) {
                const float p = __expf(s[nt][r] - m[r]);
                psum[r] += p;
                Pw[(rowhi + r) * 72 + nt * 16 + (lane & 15)] = (bf16)p;
            }
        }
#pragma unroll
        for (int r = 0; r < 8; ++r) {
            float t = psum[r];
#pragma unroll
            for (int d = 1; d < 16; d <<= 1) t += __shfl_xor(t, d, 32);
            lsum[r] += t;
        }

        // O += P @ V  (wave-local LDS round trip; same-wave LDS ops are in-order)
#pragma unroll
        for (int c = 0; c < 2; ++c) {
            const v16bf pa = load_a_frag(Pw, 0, c * 32, 72, lane);
#pragma unroll
            for (int dt = 0; dt < 4; ++dt) {
                const v16bf vb = load_b_frag(Vs, c * 32, dt * 16, 64, lane);
                o[dt] = wmma_bf16(pa, vb, o[dt]);
            }
        }
    }

    // normalize and scatter into concat-heads layout: SA[b][row][h*64+col]
#pragma unroll
    for (int dt = 0; dt < 4; ++dt) {
        const int col = dt * 16 + (lane & 15);
#pragma unroll
        for (int r = 0; r < 8; ++r) {
            const int row = q0 + wave * 16 + rowhi + r;
            const float v = o[dt][r] / lsum[r];
            SA[((size_t)b * NSEQ + row) * DMODEL + h * HD + col] = (bf16)v;
        }
    }
}

// ---------------------------------------------------------------------------
// Kernel 3: output projection  out = SA @ Wo + bo  (f32 out)
// ---------------------------------------------------------------------------
__global__ __launch_bounds__(256)
void proj_kernel(const bf16* __restrict__ SA, const float* __restrict__ Wo,
                 const float* __restrict__ bo, float* __restrict__ out) {
    const int mt  = blockIdx.x;         // 128-row tile: 0..31
    const int nt0 = blockIdx.y * 128;   // 128-col tile: 8 tiles
    const int tid = threadIdx.x, lane = tid & 31, wave = tid >> 5;

    __shared__ bf16 As[128 * 32];
    __shared__ bf16 Bs[32 * 128];

    v8f acc[8];
#pragma unroll
    for (int t = 0; t < 8; ++t) acc[t] = (v8f){0, 0, 0, 0, 0, 0, 0, 0};

    const bf16* Ab = SA + (size_t)mt * 128 * DMODEL;

    for (int kk = 0; kk < DMODEL; kk += 32) {
        __builtin_prefetch(Wo + (size_t)(kk + 32) * DMODEL + nt0, 0, 0);
        // A tile: dword-granularity copy (bf16 pairs)
        for (int idx = tid; idx < 2048; idx += 256) {
            const int r = idx >> 4, cu = idx & 15;  // 16 dwords per 32-elem row
            ((unsigned*)As)[idx] =
                ((const unsigned*)(Ab + (size_t)r * DMODEL + kk))[cu];
        }
        for (int idx = tid; idx < 32 * 128; idx += 256) {
            const int r = idx >> 7, c = idx & 127;
            Bs[idx] = (bf16)Wo[(size_t)(kk + r) * DMODEL + nt0 + c];
        }
        __syncthreads();
        const v16bf a = load_a_frag(As, wave * 16, 0, 32, lane);
#pragma unroll
        for (int t = 0; t < 8; ++t) {
            const v16bf bf = load_b_frag(Bs, 0, t * 16, 128, lane);
            acc[t] = wmma_bf16(a, bf, acc[t]);
        }
        __syncthreads();
    }

    const int rowhi = (lane >> 4) << 3;
#pragma unroll
    for (int t = 0; t < 8; ++t) {
        const int col = nt0 + t * 16 + (lane & 15);
        const float bias = bo[col];
#pragma unroll
        for (int r = 0; r < 8; ++r) {
            const int row = mt * 128 + wave * 16 + rowhi + r;
            out[(size_t)row * DMODEL + col] = acc[t][r] + bias;
        }
    }
}

// ---------------------------------------------------------------------------
extern "C" void kernel_launch(void* const* d_in, const int* in_sizes, int n_in,
                              void* d_out, int out_size, void* d_ws, size_t ws_size,
                              hipStream_t stream) {
    (void)in_sizes; (void)n_in; (void)out_size; (void)ws_size;
    const float* x    = (const float*)d_in[0];
    const float* Wqkv = (const float*)d_in[1];
    const float* bqkv = (const float*)d_in[2];
    const float* Wo   = (const float*)d_in[3];
    const float* bo   = (const float*)d_in[4];
    float* out = (float*)d_out;

    const size_t perTensor = (size_t)BH * NSEQ * HD;  // 4,194,304 bf16 elems
    bf16* Kb = (bf16*)d_ws;
    bf16* Qb = Kb + perTensor;
    bf16* Vb = Qb + perTensor;
    bf16* SA = Vb + perTensor;  // [B][N][D] bf16, 8 MB; total ws use = 32 MB

    qkv_kernel<<<dim3(BH, NSEQ / 64), 256, 0, stream>>>(x, Wqkv, bqkv, Kb, Qb, Vb);
    flash_kernel<<<dim3(BH, NSEQ / 64), 128, 0, stream>>>(Qb, Kb, Vb, SA);
    proj_kernel<<<dim3((BATCH * NSEQ) / 128, DMODEL / 128), 256, 0, stream>>>(SA, Wo, bo, out);
}
